// MultiHeadAttentionLayer_79791902425475
// MI455X (gfx1250) — compile-verified, matrix operand order checked by reading, not run on hardware
//
#include <hip/hip_runtime.h>

// ---------------------------------------------------------------------------
// MHA forward for MI455X (gfx1250): bf16 WMMA + async global->LDS staging.
// B=2, S=2048, D=1024, H=16, HD=64.
// ---------------------------------------------------------------------------

#define BATCH 2
#define S_LEN 2048
#define DMODEL 1024
#define NHEAD 16
#define HDIM 64
#define ROWS (BATCH * S_LEN)          // 4096 token rows
#define KSTEPS (DMODEL / 32)          // 32 k-steps per GEMM

typedef __attribute__((ext_vector_type(16))) __bf16 v16bf;
typedef __attribute__((ext_vector_type(8)))  __bf16 v8bf;   // 16-byte vector
typedef __attribute__((ext_vector_type(8)))  float  v8f;

__device__ __forceinline__ v8f wmma_bf16(v16bf a, v16bf b, v8f c) {
  return __builtin_amdgcn_wmma_f32_16x16x32_bf16(
      false, a, false, b, (short)0, c, false, false);
}

// Async DMA: 16 bytes per lane, global -> LDS, tracked by ASYNCcnt.
__device__ __forceinline__ void async_load_b128(const __bf16* gsrc, __bf16* ldst) {
  unsigned lo = (unsigned)(unsigned long long)ldst;   // LDS byte address
  unsigned long long ga = (unsigned long long)gsrc;
  asm volatile("global_load_async_to_lds_b128 %0, %1, off"
               :: "v"(lo), "v"(ga) : "memory");
}
__device__ __forceinline__ void wait_async_le1() {
  asm volatile("s_wait_asynccnt 1" ::: "memory");
}

// 16-bit fragment, contraction dim CONTIGUOUS in memory: elem(k)=p0[rc*ld+k].
// Lanes<16 hold k {0..7,16..23}; lanes>=16 shifted +8 (ISA 7.12.2).
__device__ __forceinline__ v16bf ld_frag_kcontig(const __bf16* p0, int ld,
                                                 int rc, int kk, int lane) {
  int half = (lane >> 4) & 1;
  const __bf16* p = p0 + (size_t)rc * ld + kk + 8 * half;
  v8bf lo = *(const v8bf*)(p);
  v8bf hi = *(const v8bf*)(p + 16);
  v16bf f;
#pragma unroll
  for (int i = 0; i < 8; ++i) { f[i] = lo[i]; f[8 + i] = hi[i]; }
  return f;
}

// ---------------------------------------------------------------------------
__global__ void cvt_f32_bf16(const float* __restrict__ in,
                             __bf16* __restrict__ out, int n) {
  int i = blockIdx.x * blockDim.x + threadIdx.x;
  if (i < n) out[i] = (__bf16)in[i];
}

// W[K,N] f32 -> WT[N,K] bf16 (one-time; makes GEMM B-frags k-contiguous)
__global__ void cvt_transpose(const float* __restrict__ in,
                              __bf16* __restrict__ out, int K, int N) {
  int i = blockIdx.x * blockDim.x + threadIdx.x;
  if (i < K * N) {
    int k = i / N, n = i - k * N;
    out[(size_t)n * K + k] = (__bf16)in[i];
  }
}

// ---------------------------------------------------------------------------
// C[4096,1024] = A[4096,1024] * WT[1024,1024]^T + bias.
// 256 thr = 8 waves; wave w owns rows [m0,m0+16) x 64 cols.
// Weight tile (64x32) double-buffered in LDS via async global->LDS DMA.
// Branch-free 2x-unrolled pipeline: prefetch tile s+1 / A-frag s+1 while
// consuming tile s. One phantom prefetch past the end (lands in adjacent
// workspace buffers; never consumed) keeps the loop uniform.
// transV: write bf16 output transposed per-batch as [B, D, S] (for V).
__global__ __launch_bounds__(256) void gemm_bias(
    const __bf16* __restrict__ A, const __bf16* __restrict__ WT,
    const float* __restrict__ bias, __bf16* __restrict__ outb,
    float* __restrict__ outf, int transV) {
  __shared__ __align__(16) __bf16 bt[2][64 * 32];   // 2 x 4KB tiles

  int lane = threadIdx.x & 31;
  int wave = threadIdx.x >> 5;
  int l15 = lane & 15;
  int half = (lane >> 4) & 1;
  int m0 = blockIdx.y * 128 + wave * 16;
  int n0 = blockIdx.x * 64;

  int tn = threadIdx.x >> 2;            // 0..63 : column within tile
  int tk = (threadIdx.x & 3) * 8;       // 0,8,16,24 : k within step

  const __bf16* wsrc = WT + (size_t)(n0 + tn) * DMODEL + tk;
  __bf16* lds0 = &bt[0][tn * 32 + tk];
  __bf16* lds1 = &bt[1][tn * 32 + tk];
  int arow = m0 + l15;

  // prologue
  async_load_b128(wsrc, lds0);                       // tile 0
  v16bf a0 = ld_frag_kcontig(A, DMODEL, arow, 0, lane);

  v8f acc[4] = {};
  for (int s = 0; s < KSTEPS; s += 2) {
    int kk = s * 32;
    // ---- half 1: consume (a0, buf0); prefetch tile s+1 ----
    async_load_b128(wsrc + (kk + 32), lds1);
    v16bf a1 = ld_frag_kcontig(A, DMODEL, arow, kk + 32, lane);
    wait_async_le1();            // tile s complete; tile s+1 in flight
    __syncthreads();
#pragma unroll
    for (int j = 0; j < 4; ++j) {
      v16bf b = ld_frag_kcontig(bt[0], 32, j * 16 + l15, 0, lane);
      acc[j] = wmma_bf16(a0, b, acc[j]);
    }
    __syncthreads();
    // ---- half 2: consume (a1, buf1); prefetch tile s+2 (phantom-safe) ----
    async_load_b128(wsrc + (kk + 64), lds0);
    a0 = ld_frag_kcontig(A, DMODEL, arow, kk + 64, lane);
    wait_async_le1();            // tile s+1 complete; tile s+2 in flight
    __syncthreads();
#pragma unroll
    for (int j = 0; j < 4; ++j) {
      v16bf b = ld_frag_kcontig(bt[1], 32, j * 16 + l15, 0, lane);
      acc[j] = wmma_bf16(a1, b, acc[j]);
    }
    __syncthreads();
  }

#pragma unroll
  for (int j = 0; j < 4; ++j) {
    int col = n0 + j * 16 + l15;
    float bb = bias[col];
#pragma unroll
    for (int r = 0; r < 8; ++r) {
      int row = m0 + r + 8 * half;
      float v = acc[j][r] + bb;
      if (outb) {
        size_t idx;
        if (transV) {   // [B, D, S] layout for attention V fragments
          int b_ = row / S_LEN, s_ = row - b_ * S_LEN;
          idx = ((size_t)b_ * DMODEL + col) * S_LEN + s_;
        } else {
          idx = (size_t)row * DMODEL + col;
        }
        outb[idx] = (__bf16)v;
      } else {
        outf[(size_t)row * DMODEL + col] = v;
      }
    }
  }
}

// ---------------------------------------------------------------------------
// Flash attention: one wave per (16-query tile, head, batch).
// scores = Q K^T / 8, online softmax, ctx = P V. All matmuls via WMMA bf16.
// q,k: [B*S, D] bf16; vT: [B, D, S] bf16 (k-contiguous for P*V frags).
__global__ __launch_bounds__(32) void attn_kernel(
    const __bf16* __restrict__ qb, const __bf16* __restrict__ kb,
    const __bf16* __restrict__ vT, __bf16* __restrict__ ctxb) {
  __shared__ __align__(16) __bf16 pls[16 * 32];  // P: C-layout -> A-layout

  int lane = threadIdx.x;
  int l15 = lane & 15;
  int half = (lane >> 4) & 1;
  int qt = blockIdx.x;
  int h  = blockIdx.y;
  int b  = blockIdx.z;

  const __bf16* qtile = qb + (size_t)(b * S_LEN + qt * 16) * DMODEL + h * HDIM;
  const __bf16* kbase = kb + (size_t)(b * S_LEN) * DMODEL + h * HDIM;
  const __bf16* vhead = vT + ((size_t)b * DMODEL + h * HDIM) * S_LEN;

  v16bf q0 = ld_frag_kcontig(qtile, DMODEL, l15, 0,  lane);
  v16bf q1 = ld_frag_kcontig(qtile, DMODEL, l15, 32, lane);

  float rmax[8], rsum[8];
  v8f cacc[4] = {};
#pragma unroll
  for (int r = 0; r < 8; ++r) { rmax[r] = -1.0e30f; rsum[r] = 0.0f; }

  for (int kc = 0; kc < S_LEN; kc += 32) {
    v8f s0 = {}, s1 = {};
    {
      v16bf k0a = ld_frag_kcontig(kbase + (size_t)(kc)      * DMODEL, DMODEL, l15, 0,  lane);
      v16bf k0b = ld_frag_kcontig(kbase + (size_t)(kc)      * DMODEL, DMODEL, l15, 32, lane);
      v16bf k1a = ld_frag_kcontig(kbase + (size_t)(kc + 16) * DMODEL, DMODEL, l15, 0,  lane);
      v16bf k1b = ld_frag_kcontig(kbase + (size_t)(kc + 16) * DMODEL, DMODEL, l15, 32, lane);
      s0 = wmma_bf16(q0, k0a, s0); s0 = wmma_bf16(q1, k0b, s0);
      s1 = wmma_bf16(q0, k1a, s1); s1 = wmma_bf16(q1, k1b, s1);
    }

    // online softmax: C-layout row = r + 8*half, col across 16 lanes
    float p0[8], p1[8];
#pragma unroll
    for (int r = 0; r < 8; ++r) {
      float a = s0[r] * 0.125f;
      float c = s1[r] * 0.125f;
      float m = fmaxf(a, c);
      m = fmaxf(m, __shfl_xor(m, 1, 32));
      m = fmaxf(m, __shfl_xor(m, 2, 32));
      m = fmaxf(m, __shfl_xor(m, 4, 32));
      m = fmaxf(m, __shfl_xor(m, 8, 32));
      float nm = fmaxf(rmax[r], m);
      float sc = __expf(rmax[r] - nm);
      rmax[r] = nm;
      p0[r] = __expf(a - nm);
      p1[r] = __expf(c - nm);
      float ps = p0[r] + p1[r];
      ps += __shfl_xor(ps, 1, 32);
      ps += __shfl_xor(ps, 2, 32);
      ps += __shfl_xor(ps, 4, 32);
      ps += __shfl_xor(ps, 8, 32);
      rsum[r] = rsum[r] * sc + ps;
#pragma unroll
      for (int j = 0; j < 4; ++j) cacc[j][r] *= sc;
    }

    // P: C-layout -> A-frag via LDS
    __syncthreads();
#pragma unroll
    for (int r = 0; r < 8; ++r) {
      pls[(r + 8 * half) * 32 + l15]      = (__bf16)p0[r];
      pls[(r + 8 * half) * 32 + 16 + l15] = (__bf16)p1[r];
    }
    __syncthreads();
    v16bf pA = ld_frag_kcontig(pls, 32, l15, 0, lane);

#pragma unroll
    for (int j = 0; j < 4; ++j) {
      v16bf vB = ld_frag_kcontig(vhead, S_LEN, j * 16 + l15, kc, lane);
      cacc[j] = wmma_bf16(pA, vB, cacc[j]);
    }
    __syncthreads();
  }

  __bf16* op = ctxb + (size_t)(b * S_LEN + qt * 16) * DMODEL + h * HDIM;
#pragma unroll
  for (int j = 0; j < 4; ++j) {
#pragma unroll
    for (int r = 0; r < 8; ++r) {
      op[(size_t)(r + 8 * half) * DMODEL + j * 16 + l15] =
          (__bf16)(cacc[j][r] / rsum[r]);
    }
  }
}

// ---------------------------------------------------------------------------
extern "C" void kernel_launch(void* const* d_in, const int* in_sizes, int n_in,
                              void* d_out, int out_size, void* d_ws, size_t ws_size,
                              hipStream_t stream) {
  const float* x  = (const float*)d_in[0];
  const float* Wq = (const float*)d_in[1];
  const float* bq = (const float*)d_in[2];
  const float* Wk = (const float*)d_in[3];
  const float* bk = (const float*)d_in[4];
  const float* Wv = (const float*)d_in[5];
  const float* bv = (const float*)d_in[6];
  const float* Wo = (const float*)d_in[7];
  const float* bo = (const float*)d_in[8];

  const size_t ND = (size_t)ROWS * DMODEL;    // 4M elems
  const size_t WD = (size_t)DMODEL * DMODEL;  // 1M elems

  __bf16* ws  = (__bf16*)d_ws;
  __bf16* xb  = ws;                  // x  bf16 [4096,1024]
  __bf16* wqT = xb  + ND;            // Wq^T bf16 [1024,1024]
  __bf16* wkT = wqT + WD;
  __bf16* wvT = wkT + WD;
  __bf16* woT = wvT + WD;
  __bf16* qbf = woT + WD;            // Q [4096,1024]
  __bf16* kbf = qbf + ND;            // K [4096,1024]
  __bf16* vTt = kbf + ND;            // V^T [B,1024,2048]
  __bf16* ctx = vTt + ND;            // ctx [4096,1024]

  {
    int n = (int)ND;
    cvt_f32_bf16<<<(n + 255) / 256, 256, 0, stream>>>(x, xb, n);
    int w = (int)WD;
    cvt_transpose<<<(w + 255) / 256, 256, 0, stream>>>(Wq, wqT, DMODEL, DMODEL);
    cvt_transpose<<<(w + 255) / 256, 256, 0, stream>>>(Wk, wkT, DMODEL, DMODEL);
    cvt_transpose<<<(w + 255) / 256, 256, 0, stream>>>(Wv, wvT, DMODEL, DMODEL);
    cvt_transpose<<<(w + 255) / 256, 256, 0, stream>>>(Wo, woT, DMODEL, DMODEL);
  }

  dim3 ggrid(DMODEL / 64, ROWS / 128);
  gemm_bias<<<ggrid, 256, 0, stream>>>(xb, wqT, bq, qbf, nullptr, 0);
  gemm_bias<<<ggrid, 256, 0, stream>>>(xb, wkT, bk, kbf, nullptr, 0);
  gemm_bias<<<ggrid, 256, 0, stream>>>(xb, wvT, bv, vTt, nullptr, 1);

  attn_kernel<<<dim3(S_LEN / 16, NHEAD, BATCH), 32, 0, stream>>>(qbf, kbf, vTt, ctx);

  gemm_bias<<<ggrid, 256, 0, stream>>>(ctx, woT, bo, nullptr, (float*)d_out, 0);
}
